// Block_7645041787038
// MI455X (gfx1250) — compile-verified, matrix operand order checked by reading, not run on hardware
//
#include <hip/hip_runtime.h>

// ============================================================================
// ViT transformer block for MI455X (gfx1250, wave32, WMMA).
//
// Strategy: all GEMMs and attention matmuls run through
// v_wmma_f32_16x16x32_f16 (f16 operands, fp32 accumulate). fp32 inputs /
// weights are staged to f16 once; every epilogue (bias, residual, exact GELU,
// softmax scramble-store) is fused. Attention uses a flash-style online
// softmax per 16-query wave tile with a 1KB per-wave LDS transpose buffer to
// move P from C-fragment layout to A-fragment layout.
//
// Workspace requirement: ~215 MB (asserted only by construction; layout below).
// ============================================================================

typedef __attribute__((ext_vector_type(16))) _Float16 v16h;
typedef __attribute__((ext_vector_type(8)))  float    v8f;

#define Bb   8
#define Nn   2049
#define Cc   768
#define Hh   12
#define HD   64
#define NBr  4
#define SPB  512
#define BTOT (Bb * Nn)          // 16392 rows
#define SCALE_ATT 0.125f        // 64^-0.5
#define QKV_LD 2304

union F16x16 { v16h v; _Float16 h[16]; };

// ---------------------------------------------------------------------------
// Fragment loaders for the CDNA5 16-bit 16x32 WMMA operand layout:
//   lane L holds matrix row (A) / column (B) = L&15.
//   half index i -> K = (i<8 ? kb+i : 16+kb+i-8), kb = (L&16)?8:0.
// Both halves are 8 contiguous f16 = one 16B load each.
// ---------------------------------------------------------------------------
__device__ inline v16h frag_rowmajor(const _Float16* base, int ld, int lane) {
  const int row = lane & 15;
  const int kb  = (lane & 16) ? 8 : 0;
  const _Float16* p = base + (size_t)row * ld;
  F16x16 u;
#pragma unroll
  for (int i = 0; i < 8; ++i) u.h[i] = p[kb + i];
#pragma unroll
  for (int i = 0; i < 8; ++i) u.h[8 + i] = p[16 + kb + i];
  return u.v;
}

__device__ inline v16h frag_rowmajor_m(const _Float16* base, int ld, int nrows, int lane) {
  const int row = lane & 15;
  const int kb  = (lane & 16) ? 8 : 0;
  F16x16 u;
#pragma unroll
  for (int i = 0; i < 16; ++i) u.h[i] = (_Float16)0.f;
  if (row < nrows) {
    const _Float16* p = base + (size_t)row * ld;
#pragma unroll
    for (int i = 0; i < 8; ++i) u.h[i] = p[kb + i];
#pragma unroll
    for (int i = 0; i < 8; ++i) u.h[8 + i] = p[16 + kb + i];
  }
  return u.v;
}

// B operand where memory is [K rows x cols] row-major and fragment wants
// column (lane&15) across K: strided gather (used for V in attention).
__device__ inline v16h frag_colmajor(const _Float16* base, int ld, int lane) {
  const int col = lane & 15;
  const int kb  = (lane & 16) ? 8 : 0;
  F16x16 u;
#pragma unroll
  for (int i = 0; i < 8; ++i) u.h[i] = base[(size_t)(kb + i) * ld + col];
#pragma unroll
  for (int i = 0; i < 8; ++i) u.h[8 + i] = base[(size_t)(16 + kb + i) * ld + col];
  return u.v;
}

__device__ inline v8f wmma_f16(v16h a, v16h b, v8f c) {
  // D = A(16x32) * B(32x16) + C, fp32 accumulate
  return __builtin_amdgcn_wmma_f32_16x16x32_f16(
      /*neg_a=*/false, a, /*neg_b=*/false, b,
      /*c_mod=*/(short)0, c, /*reuse_a=*/false, /*reuse_b=*/false);
}

// ---------------------------------------------------------------------------
// fp32 -> f16 staging
// ---------------------------------------------------------------------------
__global__ __launch_bounds__(256) void f32_to_f16_kernel(
    const float* __restrict__ src, _Float16* __restrict__ dst, int n) {
  int i = blockIdx.x * 256 + threadIdx.x;
  if (i < n) dst[i] = (_Float16)src[i];
}

// ---------------------------------------------------------------------------
// LayerNorm over C=768 (one block per row). Writes f16 activations; when
// cls_out != nullptr also writes fp32 row-0-of-each-batch (for cls path).
// ---------------------------------------------------------------------------
__global__ __launch_bounds__(256) void layernorm_kernel(
    const float* __restrict__ X, const float* __restrict__ g,
    const float* __restrict__ bta, _Float16* __restrict__ Y16,
    float* __restrict__ cls_out) {
  const int row = blockIdx.x;
  const float* x = X + (size_t)row * Cc;
  __shared__ float red[256];
  float v[3];
  float s = 0.f;
#pragma unroll
  for (int i = 0; i < 3; ++i) { v[i] = x[threadIdx.x + 256 * i]; s += v[i]; }
  red[threadIdx.x] = s; __syncthreads();
  for (int st = 128; st > 0; st >>= 1) {
    if (threadIdx.x < st) red[threadIdx.x] += red[threadIdx.x + st];
    __syncthreads();
  }
  const float mu = red[0] * (1.0f / Cc);
  __syncthreads();
  float s2 = 0.f;
#pragma unroll
  for (int i = 0; i < 3; ++i) { float d = v[i] - mu; s2 += d * d; }
  red[threadIdx.x] = s2; __syncthreads();
  for (int st = 128; st > 0; st >>= 1) {
    if (threadIdx.x < st) red[threadIdx.x] += red[threadIdx.x + st];
    __syncthreads();
  }
  const float rstd = rsqrtf(red[0] * (1.0f / Cc) + 1e-6f);
#pragma unroll
  for (int i = 0; i < 3; ++i) {
    const int c = threadIdx.x + 256 * i;
    const float o = (v[i] - mu) * rstd * g[c] + bta[c];
    Y16[(size_t)row * Cc + c] = (_Float16)o;
    if (cls_out != nullptr && (row % Nn) == 0)
      cls_out[(size_t)(row / Nn) * Cc + c] = o;
  }
}

// ---------------------------------------------------------------------------
// Generic WMMA GEMM: out[m, j] = sum_k A[m,k] * W[j,k]  (both K-fast, f16)
// Wave computes a 16x64 tile of C. EPI: 0 = store f16
//                                       1 = +bias +resid, store f32
//                                       2 = +bias, exact GELU, store f16
// ---------------------------------------------------------------------------
template <int EPI>
__global__ __launch_bounds__(256) void gemm16x64_kernel(
    const _Float16* __restrict__ A, int lda,
    const _Float16* __restrict__ W, int ldw,
    int M, int K,
    const float* __restrict__ bias, const float* __restrict__ resid,
    float* __restrict__ outf, _Float16* __restrict__ outh, int ldo) {
  const int lane = threadIdx.x & 31;
  const int wave = threadIdx.x >> 5;
  const int mt = blockIdx.y * 8 + wave;
  const int m0 = mt * 16;
  if (m0 >= M) return;                     // wave-uniform exit
  const int j0 = blockIdx.x * 64;
  const int nrows = M - m0;
  const bool full = (nrows >= 16);

  v8f acc0 = {}; v8f acc1 = {}; v8f acc2 = {}; v8f acc3 = {};
  const _Float16* Ab = A + (size_t)m0 * lda;

  for (int k0 = 0; k0 < K; k0 += 32) {
    __builtin_prefetch(Ab + k0 + 64, 0, 1);
    v16h a = full ? frag_rowmajor(Ab + k0, lda, lane)
                  : frag_rowmajor_m(Ab + k0, lda, nrows, lane);
    v16h b0 = frag_rowmajor(W + (size_t)(j0 +  0) * ldw + k0, ldw, lane);
    v16h b1 = frag_rowmajor(W + (size_t)(j0 + 16) * ldw + k0, ldw, lane);
    v16h b2 = frag_rowmajor(W + (size_t)(j0 + 32) * ldw + k0, ldw, lane);
    v16h b3 = frag_rowmajor(W + (size_t)(j0 + 48) * ldw + k0, ldw, lane);
    acc0 = wmma_f16(a, b0, acc0);
    acc1 = wmma_f16(a, b1, acc1);
    acc2 = wmma_f16(a, b2, acc2);
    acc3 = wmma_f16(a, b3, acc3);
  }

  const int moff = (lane >> 4) << 3;       // C-frag: lanes 16-31 hold M=r+8
  const int nn = lane & 15;
#pragma unroll
  for (int t = 0; t < 4; ++t) {
    v8f acc = (t == 0) ? acc0 : (t == 1) ? acc1 : (t == 2) ? acc2 : acc3;
    const int col = j0 + 16 * t + nn;
#pragma unroll
    for (int r = 0; r < 8; ++r) {
      const int row = m0 + moff + r;
      if (row >= M) continue;
      const float v = acc[r];
      if constexpr (EPI == 0) {
        outh[(size_t)row * ldo + col] = (_Float16)v;
      } else if constexpr (EPI == 1) {
        outf[(size_t)row * ldo + col] =
            v + bias[col] + resid[(size_t)row * ldo + col];
      } else {
        const float t2 = v + bias[col];
        const float ge = 0.5f * t2 * (1.0f + erff(t2 * 0.70710678118654752f));
        outh[(size_t)row * ldo + col] = (_Float16)ge;
      }
    }
  }
}

// ---------------------------------------------------------------------------
// cls token: q(row 0) attends over all N=2049 keys. One block per (b, head).
// Writes cls value (attn + h_row0) to fp32 buffer and to y16 row 0.
// ---------------------------------------------------------------------------
__global__ __launch_bounds__(256) void cls_attn_kernel(
    const _Float16* __restrict__ qkv, const float* __restrict__ cls_h,
    float* __restrict__ cls_val, _Float16* __restrict__ y16) {
  const int b = blockIdx.x / Hh, hh = blockIdx.x % Hh;
  __shared__ float qs[HD];
  __shared__ float sc[Nn];
  __shared__ float red[256];
  const _Float16* base = qkv + ((size_t)b * Nn) * QKV_LD + hh * HD;
  if (threadIdx.x < HD) qs[threadIdx.x] = (float)base[threadIdx.x];
  __syncthreads();
  const _Float16* kb = base + Cc;          // k part
  for (int k = threadIdx.x; k < Nn; k += 256) {
    const _Float16* kr = kb + (size_t)k * QKV_LD;
    float d = 0.f;
#pragma unroll 8
    for (int c = 0; c < HD; ++c) d += qs[c] * (float)kr[c];
    sc[k] = d * SCALE_ATT;
  }
  __syncthreads();
  float lm = -1e30f;
  for (int k = threadIdx.x; k < Nn; k += 256) lm = fmaxf(lm, sc[k]);
  red[threadIdx.x] = lm; __syncthreads();
  for (int st = 128; st > 0; st >>= 1) {
    if (threadIdx.x < st) red[threadIdx.x] = fmaxf(red[threadIdx.x], red[threadIdx.x + st]);
    __syncthreads();
  }
  const float m = red[0]; __syncthreads();
  float ls = 0.f;
  for (int k = threadIdx.x; k < Nn; k += 256) { float e = expf(sc[k] - m); sc[k] = e; ls += e; }
  red[threadIdx.x] = ls; __syncthreads();
  for (int st = 128; st > 0; st >>= 1) {
    if (threadIdx.x < st) red[threadIdx.x] += red[threadIdx.x + st];
    __syncthreads();
  }
  const float inv = 1.0f / red[0];
  if (threadIdx.x < HD) {
    const int d = threadIdx.x;
    const _Float16* vb = base + 2 * Cc + d;
    float acc = 0.f;
    for (int k = 0; k < Nn; ++k) acc += sc[k] * (float)vb[(size_t)k * QKV_LD];
    const int c = hh * HD + d;
    const float cls = acc * inv + cls_h[(size_t)b * Cc + c];
    cls_val[(size_t)b * Cc + c] = cls;
    y16[((size_t)b * Nn) * Cc + c] = (_Float16)cls;
  }
}

// cls-token second QKV: kc/vc = cls @ qkv_w.T (k and v parts only, fp32 dot).
__global__ __launch_bounds__(256) void qkvc_kernel(
    const float* __restrict__ cls_val, const float* __restrict__ qkv_w,
    _Float16* __restrict__ kc, _Float16* __restrict__ vc) {
  const int idx = blockIdx.x * 256 + threadIdx.x;
  if (idx >= Bb * 2 * Cc) return;
  const int b = idx / (2 * Cc), jj = idx % (2 * Cc);
  const float* wrow = qkv_w + (size_t)(Cc + jj) * Cc;
  const float* cb = cls_val + (size_t)b * Cc;
  float d = 0.f;
  for (int c = 0; c < Cc; ++c) d += cb[c] * wrow[c];
  if (jj < Cc) kc[(size_t)b * Cc + jj] = (_Float16)d;
  else         vc[(size_t)b * Cc + (jj - Cc)] = (_Float16)d;
}

// ---------------------------------------------------------------------------
// Branched local attention, flash-style. One wave = 16 query rows of one
// (b, head, branch). 512 keys via WMMA (32/iter) + the single cls key via
// scalar dots. Output stored with the reference's raw reshape permutation.
// ---------------------------------------------------------------------------
__global__ __launch_bounds__(256) void branch_attn_kernel(
    const _Float16* __restrict__ qkv,
    const _Float16* __restrict__ kc, const _Float16* __restrict__ vc,
    _Float16* __restrict__ y16) {
  __shared__ _Float16 plds[8][16 * 32];
  const int lane = threadIdx.x & 31;
  const int wave = threadIdx.x >> 5;
  int wid = blockIdx.x * 8 + wave;         // 0 .. 12287
  const int qt = wid & 31; wid >>= 5;      // 32 query tiles of 16
  const int nb = wid & 3;  wid >>= 2;      // branch
  const int hh = wid % Hh;
  const int b  = wid / Hh;

  const int nq0 = 1 + nb * SPB + qt * 16;
  const int nk0 = 1 + nb * SPB;
  const _Float16* qbase = qkv + ((size_t)(b * Nn + nq0)) * QKV_LD + hh * HD;
  const _Float16* kbase = qkv + ((size_t)(b * Nn + nk0)) * QKV_LD + Cc + hh * HD;
  const _Float16* vbase = qkv + ((size_t)(b * Nn + nk0)) * QKV_LD + 2 * Cc + hh * HD;

  const v16h qa0 = frag_rowmajor(qbase, QKV_LD, lane);
  const v16h qa1 = frag_rowmajor(qbase + 32, QKV_LD, lane);

  float m[8], l[8];
  v8f o0 = {}; v8f o1 = {}; v8f o2 = {}; v8f o3 = {};
#pragma unroll
  for (int r = 0; r < 8; ++r) { m[r] = -1e30f; l[r] = 0.f; }

  _Float16* pw = plds[wave];
  const int moff = (lane >> 4) << 3;
  const int nn = lane & 15;

  for (int chunk = 0; chunk < 16; ++chunk) {   // 32 keys per iteration
    const _Float16* kb0 = kbase + (size_t)(chunk * 32) * QKV_LD;
    const v16h k00 = frag_rowmajor(kb0, QKV_LD, lane);
    const v16h k01 = frag_rowmajor(kb0 + 32, QKV_LD, lane);
    const v16h k10 = frag_rowmajor(kb0 + (size_t)16 * QKV_LD, QKV_LD, lane);
    const v16h k11 = frag_rowmajor(kb0 + (size_t)16 * QKV_LD + 32, QKV_LD, lane);
    v8f s0 = {}; v8f s1 = {};
    s0 = wmma_f16(qa0, k00, s0); s0 = wmma_f16(qa1, k01, s0);
    s1 = wmma_f16(qa0, k10, s1); s1 = wmma_f16(qa1, k11, s1);

    float fac[8];
#pragma unroll
    for (int r = 0; r < 8; ++r) {
      const float a0 = s0[r] * SCALE_ATT;
      const float a1 = s1[r] * SCALE_ATT;
      float tm = fmaxf(a0, a1);              // row-max across 16 lanes of half
      tm = fmaxf(tm, __shfl_xor(tm, 1, 32));
      tm = fmaxf(tm, __shfl_xor(tm, 2, 32));
      tm = fmaxf(tm, __shfl_xor(tm, 4, 32));
      tm = fmaxf(tm, __shfl_xor(tm, 8, 32));
      const float mn = fmaxf(m[r], tm);
      const float f  = expf(m[r] - mn);
      const float p0 = expf(a0 - mn);
      const float p1 = expf(a1 - mn);
      float rs = p0 + p1;
      rs += __shfl_xor(rs, 1, 32);
      rs += __shfl_xor(rs, 2, 32);
      rs += __shfl_xor(rs, 4, 32);
      rs += __shfl_xor(rs, 8, 32);
      l[r] = l[r] * f + rs;
      m[r] = mn;
      fac[r] = f;
      const int M_ = r + moff;               // C-frag -> LDS (16x32 P tile)
      pw[M_ * 32 + nn] = (_Float16)p0;
      pw[M_ * 32 + 16 + nn] = (_Float16)p1;
    }
    __syncthreads();                          // P visible for A-layout reload
    const v16h pa = frag_rowmajor(pw, 32, lane);
    const _Float16* vb0 = vbase + (size_t)(chunk * 32) * QKV_LD;
#pragma unroll
    for (int t = 0; t < 4; ++t) {
      v8f oo = (t == 0) ? o0 : (t == 1) ? o1 : (t == 2) ? o2 : o3;
#pragma unroll
      for (int r = 0; r < 8; ++r) oo[r] *= fac[r];
      const v16h vb = frag_colmajor(vb0 + t * 16, QKV_LD, lane);
      oo = wmma_f16(pa, vb, oo);
      if (t == 0) o0 = oo; else if (t == 1) o1 = oo; else if (t == 2) o2 = oo; else o3 = oo;
    }
    __syncthreads();                          // protect pw for next iteration
  }

  // Fold in the single shared cls key (kc/vc from recomputed cls-token qkv).
  {
    const _Float16* kcp = kc + (size_t)b * Cc + hh * HD;
    const _Float16* qr  = qbase + (size_t)(lane & 15) * QKV_LD;
    float d = 0.f;
#pragma unroll 8
    for (int c = 0; c < HD; ++c) d += (float)qr[c] * (float)kcp[c];
    const float sc_lane = d * SCALE_ATT;
    float fac[8], pc[8];
#pragma unroll
    for (int r = 0; r < 8; ++r) {
      const float s = __shfl(sc_lane, moff + r, 32);
      const float mn = fmaxf(m[r], s);
      fac[r] = expf(m[r] - mn);
      pc[r]  = expf(s - mn);
      l[r] = l[r] * fac[r] + pc[r];
      m[r] = mn;
    }
    const _Float16* vcp = vc + (size_t)b * Cc + hh * HD;
#pragma unroll
    for (int t = 0; t < 4; ++t) {
      const float vv = (float)vcp[t * 16 + nn];
#pragma unroll
      for (int r = 0; r < 8; ++r) {
        v8f* op = (t == 0) ? &o0 : (t == 1) ? &o1 : (t == 2) ? &o2 : &o3;
        (*op)[r] = (*op)[r] * fac[r] + pc[r] * vv;
      }
    }
  }

  // Epilogue: normalize and store with the reference's raw reshape:
  // flat = h*SPB*HD + s*HD + d  ->  (s2, c2) = (flat/768, flat%768)
#pragma unroll
  for (int t = 0; t < 4; ++t) {
    const v8f oo = (t == 0) ? o0 : (t == 1) ? o1 : (t == 2) ? o2 : o3;
#pragma unroll
    for (int r = 0; r < 8; ++r) {
      const int s = qt * 16 + moff + r;
      const int d = t * 16 + nn;
      const int flat = hh * (SPB * HD) + s * HD + d;
      const int s2 = flat / Cc, c2 = flat % Cc;
      y16[((size_t)(b * Nn + 1 + nb * SPB + s2)) * Cc + c2] =
          (_Float16)(oo[r] / l[r]);
    }
  }
}

// ===========================================================================
extern "C" void kernel_launch(void* const* d_in, const int* in_sizes, int n_in,
                              void* d_out, int out_size, void* d_ws, size_t ws_size,
                              hipStream_t stream) {
  (void)in_sizes; (void)n_in; (void)out_size; (void)ws_size;
  const float* x      = (const float*)d_in[0];
  const float* ln1_w  = (const float*)d_in[1];
  const float* ln1_b  = (const float*)d_in[2];
  const float* qkv_w  = (const float*)d_in[3];
  const float* proj_w = (const float*)d_in[4];
  const float* proj_b = (const float*)d_in[5];
  const float* ln2_w  = (const float*)d_in[6];
  const float* ln2_b  = (const float*)d_in[7];
  const float* fc1_w  = (const float*)d_in[8];
  const float* fc1_b  = (const float*)d_in[9];
  const float* fc2_w  = (const float*)d_in[10];
  const float* fc2_b  = (const float*)d_in[11];
  float* out = (float*)d_out;

  // ---- workspace layout (256B aligned), ~215 MB total -------------------
  char* ws = (char*)d_ws;
  size_t off = 0;
  auto take = [&](size_t bytes) -> void* {
    void* p = ws + off;
    off += (bytes + 255) & ~(size_t)255;
    return p;
  };
  _Float16* h16    = (_Float16*)take((size_t)BTOT * Cc * 2);        // LN1 out, reused for LN2 out
  _Float16* big16  = (_Float16*)take((size_t)BTOT * 4 * Cc * 2);    // qkv (BTOTx2304), reused for gelu (BTOTx3072)
  _Float16* y16    = (_Float16*)take((size_t)BTOT * Cc * 2);        // attention output
  float*    x2     = (float*)   take((size_t)BTOT * Cc * 4);        // x + attn residual
  float*    clsh   = (float*)   take((size_t)Bb * Cc * 4);          // LN1 row0
  float*    clsv   = (float*)   take((size_t)Bb * Cc * 4);          // cls value
  _Float16* kc16   = (_Float16*)take((size_t)Bb * Cc * 2);
  _Float16* vc16   = (_Float16*)take((size_t)Bb * Cc * 2);
  _Float16* qkvw16 = (_Float16*)take((size_t)3 * Cc * Cc * 2);
  _Float16* projw16= (_Float16*)take((size_t)Cc * Cc * 2);
  _Float16* fc1w16 = (_Float16*)take((size_t)4 * Cc * Cc * 2);
  _Float16* fc2w16 = (_Float16*)take((size_t)4 * Cc * Cc * 2);

  // ---- stage weights to f16 ---------------------------------------------
  auto cvt = [&](const float* s, _Float16* d, int n) {
    f32_to_f16_kernel<<<(n + 255) / 256, 256, 0, stream>>>(s, d, n);
  };
  cvt(qkv_w,  qkvw16,  3 * Cc * Cc);
  cvt(proj_w, projw16, Cc * Cc);
  cvt(fc1_w,  fc1w16,  4 * Cc * Cc);
  cvt(fc2_w,  fc2w16,  4 * Cc * Cc);

  const int mtiles = (BTOT + 15) / 16;           // 1025
  const dim3 blk(256);
  const dim3 gy((unsigned)((mtiles + 7) / 8));   // 129 row-blocks

  // 1) LN1
  layernorm_kernel<<<BTOT, blk, 0, stream>>>(x, ln1_w, ln1_b, h16, clsh);

  // 2) QKV GEMM: [BTOT,768] x [2304,768]^T -> f16 [BTOT,2304]
  gemm16x64_kernel<0><<<dim3(3 * Cc / 64, gy.x), blk, 0, stream>>>(
      h16, Cc, qkvw16, Cc, BTOT, Cc, nullptr, nullptr, nullptr, big16, QKV_LD);

  // 3) cls global attention -> clsv, y16 row 0
  cls_attn_kernel<<<Bb * Hh, blk, 0, stream>>>(big16, clsh, clsv, y16);

  // 4) cls second qkv -> kc/vc
  qkvc_kernel<<<(Bb * 2 * Cc + 255) / 256, blk, 0, stream>>>(clsv, qkv_w, kc16, vc16);

  // 5) branched local attention (flash, WMMA) -> y16 rows 1..N-1
  branch_attn_kernel<<<(Bb * Hh * NBr * 32) / 8, blk, 0, stream>>>(
      big16, kc16, vc16, y16);

  // 6) proj GEMM + bias + x residual -> x2 (fp32)
  gemm16x64_kernel<1><<<dim3(Cc / 64, gy.x), blk, 0, stream>>>(
      y16, Cc, projw16, Cc, BTOT, Cc, proj_b, x, x2, nullptr, Cc);

  // 7) LN2
  layernorm_kernel<<<BTOT, blk, 0, stream>>>(x2, ln2_w, ln2_b, h16, nullptr);

  // 8) fc1 GEMM + bias + exact GELU -> f16 [BTOT,3072] (reuses big16)
  gemm16x64_kernel<2><<<dim3(4 * Cc / 64, gy.x), blk, 0, stream>>>(
      h16, Cc, fc1w16, Cc, BTOT, Cc, fc1_b, nullptr, nullptr, big16, 4 * Cc);

  // 9) fc2 GEMM + bias + x2 residual -> d_out (fp32)
  gemm16x64_kernel<1><<<dim3(Cc / 64, gy.x), blk, 0, stream>>>(
      big16, 4 * Cc, fc2w16, 4 * Cc, BTOT, 4 * Cc, fc2_b, x2, out, nullptr, Cc);
}
